// _TemporalSelfAttention_83837761618277
// MI455X (gfx1250) — matrix operand order repaired and checked by reading, not compiled, not run on hardware
//
#include <hip/hip_runtime.h>
#include <hip/hip_bf16.h>
#include <cstdint>

// Problem dims (fixed by reference)
#define B_   2
#define T_   2048
#define D_   1024
#define H_   16
#define DH_  64
#define BT_  (B_ * T_)

typedef __attribute__((ext_vector_type(8)))  __bf16 v8bf;
typedef __attribute__((ext_vector_type(16))) __bf16 v16bf;
typedef __attribute__((ext_vector_type(8)))  float  v8f;
typedef __attribute__((ext_vector_type(4)))  int    i32x4;

#if __has_builtin(__builtin_amdgcn_global_load_async_to_lds_b128)
#define HAS_ASYNC_LDS 1
#else
#define HAS_ASYNC_LDS 0
#endif

// float -> bf16 round-to-nearest-even via bit manipulation (no scalar bf16 ALU)
static __device__ __forceinline__ __bf16 f2bf(float f) {
    uint32_t u = __builtin_bit_cast(uint32_t, f);
    u += 0x7FFFu + ((u >> 16) & 1u);
    uint16_t s = (uint16_t)(u >> 16);
    return __builtin_bit_cast(__bf16, s);
}

static __device__ __forceinline__ v16bf mk16(v8bf lo, v8bf hi) {
    return __builtin_shufflevector(lo, hi, 0, 1, 2, 3, 4, 5, 6, 7,
                                           8, 9, 10, 11, 12, 13, 14, 15);
}

static __device__ __forceinline__ v8f wmma_bf16(v16bf a, v16bf b, v8f c) {
    // D = A(16x32 bf16) * B(32x16 bf16) + C(16x16 f32)
    return __builtin_amdgcn_wmma_f32_16x16x32_bf16(false, a, false, b,
                                                   (short)0, c, false, false);
}

// Copy 16B global -> LDS. Async (ASYNCcnt-tracked) when the builtin exists.
// Builtin signature (from compiler diagnostic): param0 = int4 addrspace(1)*,
// so cast through uintptr_t to the exact vector-pointer types.
static __device__ __forceinline__ void cp16(const __bf16* g, __bf16* l) {
#if HAS_ASYNC_LDS
    __builtin_amdgcn_global_load_async_to_lds_b128(
        (__attribute__((address_space(1))) i32x4*)(uintptr_t)g,
        (__attribute__((address_space(3))) i32x4*)(uint32_t)(uintptr_t)l, 0, 0);
#else
    *(uint4*)l = *(const uint4*)g;
#endif
}

static __device__ __forceinline__ void async_wait0() {
#if HAS_ASYNC_LDS
#if __has_builtin(__builtin_amdgcn_s_wait_asynccnt)
    __builtin_amdgcn_s_wait_asynccnt(0);
#else
    asm volatile("s_wait_asynccnt 0" ::: "memory");
#endif
#endif
}

// ---------------------------------------------------------------------------
// Elementwise f32 -> bf16 convert (x)
// ---------------------------------------------------------------------------
__global__ __launch_bounds__(256)
void cvt_bf16_kernel(const float* __restrict__ x, __bf16* __restrict__ y, int n) {
    int i = (blockIdx.x * 256 + threadIdx.x) * 4;
    if (i + 3 < n) {
        float4 v = *(const float4*)(x + i);
        y[i + 0] = f2bf(v.x);
        y[i + 1] = f2bf(v.y);
        y[i + 2] = f2bf(v.z);
        y[i + 3] = f2bf(v.w);
    }
}

// ---------------------------------------------------------------------------
// W[K x N] f32 -> WT[N x K] bf16 (transpose+convert; coalesced writes)
// ---------------------------------------------------------------------------
__global__ __launch_bounds__(256)
void transpose_conv_kernel(const float* __restrict__ W, __bf16* __restrict__ WT,
                           int K, int N) {
    int idx = blockIdx.x * 256 + threadIdx.x;   // idx = n*K + k
    if (idx < N * K) {
        int n = idx / K;
        int k = idx - n * K;
        WT[idx] = f2bf(W[(size_t)k * N + n]);
    }
}

// ---------------------------------------------------------------------------
// GEMM: C[M x N] = ((A[MxK]bf16 * Bt[NxK]bf16) + bias) * oscale
// Block = 4 waves; each wave computes a 32(M) x 64(N) tile via 8 WMMA tiles.
// MODE 0: bf16 row-major out [M x N]
// MODE 1: bf16 transposed out [N x M] (packed 16B column stores)
// MODE 2: f32 row-major out [M x N]
// ---------------------------------------------------------------------------
template <int MODE>
__global__ __launch_bounds__(128)
void gemm_bf16_kernel(const __bf16* __restrict__ A, const __bf16* __restrict__ Bt,
                      const float* __restrict__ bias, void* __restrict__ Cout,
                      int M, int N, int K, float oscale) {
    const int lane = threadIdx.x & 31;
    const int wave = threadIdx.x >> 5;
    const int half = lane >> 4;
    const int l16  = lane & 15;
    const int m0   = blockIdx.x * 128 + wave * 32;
    const int n0   = blockIdx.y * 64;
    const int kb8  = half * 8;    // unused; A-frag also uses kd16
    const int kd16 = half * 16;   // B-frag k base per half-wave

    const __bf16* a0 = A + (size_t)(m0 + l16) * K;        // rows m0..m0+15
    const __bf16* a1 = A + (size_t)(m0 + 16 + l16) * K;   // rows m0+16..m0+31
    const __bf16* b0 = Bt + (size_t)(n0 + l16) * K;

    v8f c[2][4] = {};

    for (int k = 0; k < K; k += 32) {
        v16bf af0 = mk16(*(const v8bf*)(a0 + k + kd16),
                         *(const v8bf*)(a0 + k + kd16 + 8));
        v16bf af1 = mk16(*(const v8bf*)(a1 + k + kd16),
                         *(const v8bf*)(a1 + k + kd16 + 8));
        if (k + 32 < K) {
            __builtin_prefetch(a0 + k + 32, 0, 0);
            __builtin_prefetch(b0 + k + 32, 0, 0);
        }
#pragma unroll
        for (int nt = 0; nt < 4; ++nt) {
            v16bf bf = *(const v16bf*)(b0 + (size_t)nt * 16 * K + k + kd16);
            c[0][nt] = wmma_bf16(af0, bf, c[0][nt]);
            c[1][nt] = wmma_bf16(af1, bf, c[1][nt]);
        }
    }

#pragma unroll
    for (int nt = 0; nt < 4; ++nt) {
        const int col = n0 + nt * 16 + l16;
        const float bv = bias[col];
#pragma unroll
        for (int mt = 0; mt < 2; ++mt) {
            const int rowbase = m0 + mt * 16 + 8 * half;
            if constexpr (MODE == 1) {
                v8bf pk;
#pragma unroll
                for (int r = 0; r < 8; ++r)
                    pk[r] = f2bf((c[mt][nt][r] + bv) * oscale);
                *(v8bf*)((__bf16*)Cout + (size_t)col * M + rowbase) = pk;
            } else if constexpr (MODE == 0) {
#pragma unroll
                for (int r = 0; r < 8; ++r)
                    ((__bf16*)Cout)[(size_t)(rowbase + r) * N + col] =
                        f2bf((c[mt][nt][r] + bv) * oscale);
            } else {
#pragma unroll
                for (int r = 0; r < 8; ++r)
                    ((float*)Cout)[(size_t)(rowbase + r) * N + col] =
                        (c[mt][nt][r] + bv) * oscale;
            }
        }
    }
}

// ---------------------------------------------------------------------------
// Flash attention. Block = 4 waves sharing one (b,h); wave w owns q-tile
// q0 = qc*64 + w*16. K(32x64) and V^T(64x32) tiles for each 32-key block are
// staged into LDS once per block via async global->LDS loads and reused by
// all 4 waves. Q is pre-scaled by 1/sqrt(64) at projection time.
// Softmax denominator is accumulated as an extra WMMA (P @ ones).
// Qb,Kb: bf16 [B*T, D] (head h at col h*64); VT: bf16 [64, B*T];
// Ob: bf16 [B*T, D] = out.transpose(0,2,1,3).reshape
// ---------------------------------------------------------------------------
#define KSTR 72   // K LDS row stride (elements): 144B, conflict-free, 16B aligned
#define VSTR 40   // V/P LDS row stride (elements): 80B

__global__ __launch_bounds__(128)
void attn_kernel(const __bf16* __restrict__ Qb, const __bf16* __restrict__ Kb,
                 const __bf16* __restrict__ VT, __bf16* __restrict__ Ob) {
    __shared__ __bf16 klds[32 * KSTR];      // 32 keys x 64 dh
    __shared__ __bf16 vlds[64 * VSTR];      // 64 dh x 32 keys
    __shared__ __bf16 plds[4][16 * VSTR];   // per-wave 16x32 P tile

    const int tid  = threadIdx.x;
    const int lane = tid & 31;
    const int wave = tid >> 5;
    const int half = lane >> 4;
    const int l16  = lane & 15;
    const int qc   = blockIdx.x & (T_ / 64 - 1);      // 0..31
    const int h    = (blockIdx.x >> 5) & (H_ - 1);
    const int b    = blockIdx.x >> 9;
    const int q0   = qc * 64 + wave * 16;
    const int kb8  = half * 8;
    const int kd16 = half * 16;

    // Q fragments (A operand); Q already carries the 1/sqrt(DH) scale.
    const __bf16* qrow = Qb + (size_t)(b * T_ + q0 + l16) * D_ + h * DH_;
    v16bf qf0 = mk16(*(const v8bf*)(qrow + kd16), *(const v8bf*)(qrow + kd16 + 8));
    v16bf qf1 = mk16(*(const v8bf*)(qrow + 32 + kd16),
                     *(const v8bf*)(qrow + 32 + kd16 + 8));

    // Ones B-fragment for the row-sum WMMA
    v16bf onesf;
#pragma unroll
    for (int i = 0; i < 16; ++i)
        onesf[i] = __builtin_bit_cast(__bf16, (uint16_t)0x3F80);

    v8f o[4] = {};
    v8f lacc = {};
    float mi[8];
#pragma unroll
    for (int r = 0; r < 8; ++r) mi[r] = -1e30f;

    __bf16* pl = plds[wave];
    const __bf16* kg_base = Kb + (size_t)(b * T_) * D_ + h * DH_;
    const __bf16* vg_base = VT + (size_t)b * T_;

    // Staging chunk ids: 256 x 16B chunks per tile, 2 per thread.
    const int c0 = tid * 2, c1 = tid * 2 + 1;

    for (int kb = 0; kb < T_; kb += 32) {
        // ---- stage K and V^T tiles into LDS (async) ----
        const __bf16* kg = kg_base + (size_t)kb * D_;   // rows: keys, stride D_
        const __bf16* vg = vg_base + kb;                // rows: dh, stride BT_
        cp16(kg + (size_t)(c0 >> 3) * D_ + (c0 & 7) * 8,
             klds + (c0 >> 3) * KSTR + (c0 & 7) * 8);
        cp16(kg + (size_t)(c1 >> 3) * D_ + (c1 & 7) * 8,
             klds + (c1 >> 3) * KSTR + (c1 & 7) * 8);
        cp16(vg + (size_t)(c0 >> 2) * BT_ + (c0 & 3) * 8,
             vlds + (c0 >> 2) * VSTR + (c0 & 3) * 8);
        cp16(vg + (size_t)(c1 >> 2) * BT_ + (c1 & 3) * 8,
             vlds + (c1 >> 2) * VSTR + (c1 & 3) * 8);
        if (kb + 32 < T_) {
            __builtin_prefetch(kg + (size_t)32 * D_ + (size_t)(c0 >> 3) * D_, 0, 0);
            __builtin_prefetch(vg + 32 + (size_t)(c0 >> 2) * BT_, 0, 0);
        }
        async_wait0();
        __syncthreads();

        // ---- fragments from LDS ----
        v16bf kf00 = *(const v16bf*)(klds + l16 * KSTR + kd16);
        v16bf kf01 = *(const v16bf*)(klds + l16 * KSTR + 32 + kd16);
        v16bf kf10 = *(const v16bf*)(klds + (16 + l16) * KSTR + kd16);
        v16bf kf11 = *(const v16bf*)(klds + (16 + l16) * KSTR + 32 + kd16);
        v16bf vf[4];
#pragma unroll
        for (int nt = 0; nt < 4; ++nt)
            vf[nt] = *(const v16bf*)(vlds + (nt * 16 + l16) * VSTR + kd16);

        // ---- scores ----
        v8f s0 = {}, s1 = {};
        s0 = wmma_bf16(qf0, kf00, s0);
        s0 = wmma_bf16(qf1, kf01, s0);
        s1 = wmma_bf16(qf0, kf10, s1);
        s1 = wmma_bf16(qf1, kf11, s1);

        // ---- online softmax (row-max butterfly stays in half-wave) ----
        float p0[8], p1[8], alpha[8];
#pragma unroll
        for (int r = 0; r < 8; ++r) {
            float t = fmaxf(s0[r], s1[r]);
            t = fmaxf(t, __shfl_xor(t, 1, 32));
            t = fmaxf(t, __shfl_xor(t, 2, 32));
            t = fmaxf(t, __shfl_xor(t, 4, 32));
            t = fmaxf(t, __shfl_xor(t, 8, 32));
            float mn = fmaxf(mi[r], t);
            alpha[r] = __expf(mi[r] - mn);
            mi[r] = mn;
            p0[r] = __expf(s0[r] - mn);
            p1[r] = __expf(s1[r] - mn);
        }
#pragma unroll
        for (int r = 0; r < 8; ++r) {
            lacc[r] *= alpha[r];
#pragma unroll
            for (int nt = 0; nt < 4; ++nt) o[nt][r] *= alpha[r];
        }

        // ---- P (C layout, f32) -> LDS bf16 ----
#pragma unroll
        for (int r = 0; r < 8; ++r) {
            const int row = r + 8 * half;
            pl[row * VSTR + l16]      = f2bf(p0[r]);
            pl[row * VSTR + 16 + l16] = f2bf(p1[r]);
        }
        // One barrier: publishes P wave-internally AND fences this iteration's
        // K/V LDS reads (already in regs) from next iteration's async writes.
        __syncthreads();
        const __bf16* prow = pl + l16 * VSTR;
        v16bf pf = mk16(*(const v8bf*)(prow + kd16), *(const v8bf*)(prow + kd16 + 8));

        // ---- O += P*V ; l += P*ones (row-sum as a WMMA) ----
#pragma unroll
        for (int nt = 0; nt < 4; ++nt) o[nt] = wmma_bf16(pf, vf[nt], o[nt]);
        lacc = wmma_bf16(pf, onesf, lacc);
    }

    // Normalize and store O (bf16, [B*T, D], head-interleaved)
    float inv[8];
#pragma unroll
    for (int r = 0; r < 8; ++r) inv[r] = 1.f / lacc[r];
#pragma unroll
    for (int nt = 0; nt < 4; ++nt)
#pragma unroll
        for (int r = 0; r < 8; ++r) {
            const int row = q0 + r + 8 * half;
            const int col = h * DH_ + nt * 16 + l16;
            Ob[(size_t)(b * T_ + row) * D_ + col] = f2bf(o[nt][r] * inv[r]);
        }
}

// ---------------------------------------------------------------------------
// Host orchestration
// ---------------------------------------------------------------------------
extern "C" void kernel_launch(void* const* d_in, const int* in_sizes, int n_in,
                              void* d_out, int out_size, void* d_ws, size_t ws_size,
                              hipStream_t stream) {
    (void)in_sizes; (void)n_in; (void)out_size; (void)ws_size;
    const float* x  = (const float*)d_in[0];
    // d_in[1] = mask (all ones) -- unused
    const float* Wq = (const float*)d_in[2];
    const float* bq = (const float*)d_in[3];
    const float* Wk = (const float*)d_in[4];
    const float* bk = (const float*)d_in[5];
    const float* Wv = (const float*)d_in[6];
    const float* bv = (const float*)d_in[7];
    const float* Wo = (const float*)d_in[8];
    const float* bo = (const float*)d_in[9];
    float* out = (float*)d_out;

    // Workspace carve-up (bf16 buffers), 256B aligned
    char* ws = (char*)d_ws;
    size_t cur = 0;
    auto alloc = [&](size_t bytes) -> char* {
        char* p = ws + cur;
        cur += (bytes + 255) & ~(size_t)255;
        return p;
    };
    __bf16* xb  = (__bf16*)alloc((size_t)BT_ * D_ * 2);   // x bf16 [BT, D]
    __bf16* WqT = (__bf16*)alloc((size_t)D_ * D_ * 2);    // [D, D]
    __bf16* WkT = (__bf16*)alloc((size_t)D_ * D_ * 2);
    __bf16* WvT = (__bf16*)alloc((size_t)DH_ * D_ * 2);   // [64, D]
    __bf16* WoT = (__bf16*)alloc((size_t)D_ * D_ * 2);
    __bf16* Qb  = (__bf16*)alloc((size_t)BT_ * D_ * 2);   // [BT, D]
    __bf16* Kb  = (__bf16*)alloc((size_t)BT_ * D_ * 2);
    __bf16* VT  = (__bf16*)alloc((size_t)DH_ * BT_ * 2);  // [64, BT]
    __bf16* Ob  = (__bf16*)alloc((size_t)BT_ * D_ * 2);

    // 1) Convert x to bf16
    {
        const int n = BT_ * D_;
        cvt_bf16_kernel<<<n / (256 * 4), 256, 0, stream>>>(x, xb, n);
    }
    // 2) Transpose+convert weights
    transpose_conv_kernel<<<(D_ * D_) / 256, 256, 0, stream>>>(Wq, WqT, D_, D_);
    transpose_conv_kernel<<<(D_ * D_) / 256, 256, 0, stream>>>(Wk, WkT, D_, D_);
    transpose_conv_kernel<<<(D_ * DH_) / 256, 256, 0, stream>>>(Wv, WvT, D_, DH_);
    transpose_conv_kernel<<<(D_ * D_) / 256, 256, 0, stream>>>(Wo, WoT, D_, D_);

    // 3) Projections. Q carries the attention scale 1/sqrt(64) (exact in bf16).
    gemm_bf16_kernel<0><<<dim3(BT_ / 128, D_ / 64), 128, 0, stream>>>(
        xb, WqT, bq, (void*)Qb, BT_, D_, D_, 0.125f);
    gemm_bf16_kernel<0><<<dim3(BT_ / 128, D_ / 64), 128, 0, stream>>>(
        xb, WkT, bk, (void*)Kb, BT_, D_, D_, 1.0f);
    gemm_bf16_kernel<1><<<dim3(BT_ / 128, DH_ / 64), 128, 0, stream>>>(
        xb, WvT, bv, (void*)VT, BT_, DH_, D_, 1.0f);

    // 4) Flash attention: block = (b, h, 64-query chunk), 4 waves
    attn_kernel<<<B_ * H_ * (T_ / 64), 128, 0, stream>>>(Qb, Kb, VT, Ob);

    // 5) Output projection -> fp32 d_out
    gemm_bf16_kernel<2><<<dim3(BT_ / 128, D_ / 64), 128, 0, stream>>>(
        Ob, WoT, bo, (void*)out, BT_, D_, D_, 1.0f);
}
